// InternLM2Attention_88201448390946
// MI455X (gfx1250) — matrix operand order repaired and checked by reading, not compiled, not run on hardware
//
#include <hip/hip_runtime.h>
#include <hip/hip_bf16.h>

// ---------------------------------------------------------------------------
// InternLM2 attention block (PLoRA qkv/wo + RoPE + causal GQA flash attention
// + beacon KV-cache fill) for MI455X (gfx1250, wave32, WMMA).
//
// Strategy (see analysis): cast everything to bf16 once per call (all bf16
// operands fit in the 192MB L2), run every matmul through
// v_wmma_f32_16x16x32_bf16 with f32 accumulation, and implement the
// .at[idx].add LoRA scatters with f32 global atomics (index repeats safe).
// ---------------------------------------------------------------------------

typedef __attribute__((ext_vector_type(16))) __bf16 v16bf;
typedef __attribute__((ext_vector_type(8)))  float  v8f;
typedef __hip_bfloat16 bf16s;

#define DEV __device__ __forceinline__

static constexpr int T_SEQ   = 2048;
static constexpr int H_DIM   = 4096;
static constexpr int NHEADS  = 32;
static constexpr int NKV     = 8;
static constexpr int HD      = 128;
static constexpr int QKV_OUT = 6144;   // (32 + 2*8) * 128
static constexpr int LORA_R  = 256;
static constexpr int N_IM    = 512;
static constexpr int N_MEM   = 128;

// ---------------- WMMA fragment loaders (wave32 layouts, ISA 7.12.2) -------

// A-matrix 16x32 bf16 (MxK).  Lane l<16: row M=l, K = 0..7 (VGPR0-3) and
// 16..23 (VGPR4-7).  Lane l+16: same row, K = 8..15 and 24..31.
DEV v16bf load_frag_a(const bf16s* base, int ld) {
  const int lane = threadIdx.x & 31;
  const int row  = lane & 15;
  const int koff = (lane >> 4) * 8;
  const bf16s* p = base + (size_t)row * ld + koff;
  union { uint4 q[2]; v16bf v; } t;
  t.q[0] = *(const uint4*)(p);        // K = koff .. koff+7
  t.q[1] = *(const uint4*)(p + 16);   // K = koff+16 .. koff+23
  return t.v;
}

// B-matrix 32x16 bf16 (KxN) where B[k][n] = Wrow[n][k] (row-major source,
// K contiguous).  Lane n<16: col N=n, K=0..7 and 16..23; lane n+16: col n, K=8..15 and 24..31.
DEV v16bf load_frag_b(const bf16s* base, int ld) {
  const int lane = threadIdx.x & 31;
  const int col  = lane & 15;
  const int koff = (lane >> 4) * 8;
  const bf16s* p = base + (size_t)col * ld + koff;
  union { uint4 q[2]; v16bf v; } t;
  t.q[0] = *(const uint4*)(p);        // K = koff .. koff+7
  t.q[1] = *(const uint4*)(p + 16);   // K = koff+16 .. koff+23
  return t.v;
}

DEV v8f wmma_bf16(v16bf a, v16bf b, v8f c) {
  return __builtin_amdgcn_wmma_f32_16x16x32_bf16(
      /*neg_a=*/false, a, /*neg_b=*/false, b,
      /*c_mod=*/(short)0, c, /*reuse_a=*/false, /*reuse_b=*/false);
}

// ---------------- GEMM:  C[M,N] = A[M,K] * W[N,K]^T  ------------------------
// Block = 256 threads = 8 waves (2x4).  Wave tile 64x64, block tile 128x256.
// MODE 0: store f32 C.   MODE 1: store bf16 Cb.
// MODE 2: atomicAdd(alpha * C) scattered to rows row_map[m]  (LoRA .add).
template<int MODE>
__global__ __launch_bounds__(256)
void gemm_bf16_wmma(const bf16s* __restrict__ A, const bf16s* __restrict__ W,
                    float* __restrict__ C, bf16s* __restrict__ Cb,
                    int K, int lda, int ldw, int ldc,
                    const int* __restrict__ row_map, float alpha) {
  const int wave  = threadIdx.x >> 5;
  const int mBase = blockIdx.x * 128 + (wave >> 2) * 64;
  const int nBase = blockIdx.y * 256 + (wave & 3) * 64;

  v8f acc[4][4];
#pragma unroll
  for (int i = 0; i < 4; ++i)
#pragma unroll
    for (int j = 0; j < 4; ++j)
#pragma unroll
      for (int e = 0; e < 8; ++e) acc[i][j][e] = 0.0f;

  for (int kk = 0; kk < K; kk += 32) {
    v16bf af[4], bw[4];
#pragma unroll
    for (int i = 0; i < 4; ++i)
      af[i] = load_frag_a(A + (size_t)(mBase + i * 16) * lda + kk, lda);
#pragma unroll
    for (int j = 0; j < 4; ++j)
      bw[j] = load_frag_b(W + (size_t)(nBase + j * 16) * ldw + kk, ldw);
#pragma unroll
    for (int i = 0; i < 4; ++i)
#pragma unroll
      for (int j = 0; j < 4; ++j)
        acc[i][j] = wmma_bf16(af[i], bw[j], acc[i][j]);
  }

  // C/D layout: lane n<16 -> col n, rows r (VGPR r) = 0..7; lane n+16 -> rows 8..15
  const int lane = threadIdx.x & 31;
  const int col  = lane & 15;
  const int rb   = (lane >> 4) * 8;
#pragma unroll
  for (int i = 0; i < 4; ++i)
#pragma unroll
    for (int j = 0; j < 4; ++j)
#pragma unroll
      for (int r = 0; r < 8; ++r) {
        const int row = mBase + i * 16 + rb + r;
        const int cn  = nBase + j * 16 + col;
        const float v = acc[i][j][r];
        if (MODE == 0) {
          C[(size_t)row * ldc + cn] = v;
        } else if (MODE == 1) {
          Cb[(size_t)row * ldc + cn] = __float2bfloat16(v);
        } else {
          atomicAdd(&C[(size_t)row_map[row] * ldc + cn], v * alpha);
        }
      }
}

// ---------------- Flash attention (causal, GQA 32q/8kv, HD=128) -------------
// One wave per (16-row q block, head).  Q fragments held in registers.
// S = Q*K^T via WMMA; online softmax with shfl_xor row reductions;
// P staged in LDS as a 16x32 bf16 A-tile; O += P*V via WMMA on transposed V.
__global__ __launch_bounds__(32)
void attn_fwd(const bf16s* __restrict__ Qr, const bf16s* __restrict__ Kr,
              const bf16s* __restrict__ Vt, bf16s* __restrict__ Ao) {
  const int h    = blockIdx.y;
  const int hkv  = h >> 2;
  const int q0   = blockIdx.x * 16;
  const int lane = threadIdx.x & 31;
  const int col  = lane & 15;
  const int rb   = (lane >> 4) * 8;
  __shared__ bf16s Pt[16 * 32];

  v16bf qa[4];
#pragma unroll
  for (int kc = 0; kc < 4; ++kc)
    qa[kc] = load_frag_a(Qr + (size_t)q0 * H_DIM + h * HD + kc * 32, H_DIM);

  v8f   o[8];
  float mrun[8], lrun[8];
#pragma unroll
  for (int j = 0; j < 8; ++j)
#pragma unroll
    for (int e = 0; e < 8; ++e) o[j][e] = 0.0f;
#pragma unroll
  for (int r = 0; r < 8; ++r) { mrun[r] = -1e30f; lrun[r] = 0.0f; }

  const float sc   = 0.08838834764831845f;  // 1/sqrt(128)
  const int sbmax  = (q0 + 15) >> 5;

  for (int sb = 0; sb <= sbmax; ++sb) {
    const int s0 = sb * 32;
    v8f st[2];
#pragma unroll
    for (int t = 0; t < 2; ++t)
#pragma unroll
      for (int e = 0; e < 8; ++e) st[t][e] = 0.0f;

#pragma unroll
    for (int t = 0; t < 2; ++t)
#pragma unroll
      for (int kc = 0; kc < 4; ++kc) {
        v16bf bk = load_frag_b(
            Kr + (size_t)(s0 + t * 16) * (NKV * HD) + hkv * HD + kc * 32,
            NKV * HD);
        st[t] = wmma_bf16(qa[kc], bk, st[t]);
      }

    // scale + causal mask + row max
    float mx[8];
#pragma unroll
    for (int r = 0; r < 8; ++r) mx[r] = -1e30f;
#pragma unroll
    for (int t = 0; t < 2; ++t)
#pragma unroll
      for (int r = 0; r < 8; ++r) {
        float v = st[t][r] * sc;
        const int s    = s0 + t * 16 + col;
        const int qrow = q0 + rb + r;
        if (s > qrow) v = -1e30f;
        st[t][r] = v;
        mx[r] = fmaxf(mx[r], v);
      }
#pragma unroll
    for (int m = 1; m < 16; m <<= 1)
#pragma unroll
      for (int r = 0; r < 8; ++r) mx[r] = fmaxf(mx[r], __shfl_xor(mx[r], m, 32));

    float scl[8], rs[8];
#pragma unroll
    for (int r = 0; r < 8; ++r) {
      const float mn = fmaxf(mrun[r], mx[r]);
      scl[r]  = __expf(mrun[r] - mn);
      mrun[r] = mn;
      rs[r]   = 0.0f;
    }
#pragma unroll
    for (int t = 0; t < 2; ++t)
#pragma unroll
      for (int r = 0; r < 8; ++r) {
        const float p = __expf(st[t][r] - mrun[r]);
        st[t][r] = p;
        rs[r] += p;
      }
#pragma unroll
    for (int m = 1; m < 16; m <<= 1)
#pragma unroll
      for (int r = 0; r < 8; ++r) rs[r] += __shfl_xor(rs[r], m, 32);
#pragma unroll
    for (int r = 0; r < 8; ++r) lrun[r] = lrun[r] * scl[r] + rs[r];
#pragma unroll
    for (int j = 0; j < 8; ++j)
#pragma unroll
      for (int r = 0; r < 8; ++r) o[j][r] *= scl[r];

    // stage P (16x32) to LDS in row-major bf16, reload as A fragment
#pragma unroll
    for (int t = 0; t < 2; ++t)
#pragma unroll
      for (int r = 0; r < 8; ++r)
        Pt[(rb + r) * 32 + t * 16 + col] = __float2bfloat16(st[t][r]);
    __syncthreads();
    v16bf pa = load_frag_a(Pt, 32);
#pragma unroll
    for (int j = 0; j < 8; ++j) {
      v16bf bv = load_frag_b(
          Vt + (size_t)hkv * HD * T_SEQ + (size_t)(j * 16) * T_SEQ + s0, T_SEQ);
      o[j] = wmma_bf16(pa, bv, o[j]);
    }
    __syncthreads();
  }

#pragma unroll
  for (int r = 0; r < 8; ++r) lrun[r] = 1.0f / lrun[r];
#pragma unroll
  for (int j = 0; j < 8; ++j)
#pragma unroll
    for (int r = 0; r < 8; ++r)
      Ao[(size_t)(q0 + rb + r) * H_DIM + h * HD + j * 16 + col] =
          __float2bfloat16(o[j][r] * lrun[r]);
}

// ---------------- Elementwise helpers --------------------------------------

__global__ void cast_f32_to_bf16(const float* __restrict__ s,
                                 bf16s* __restrict__ d, long n) {
  long i = (long)blockIdx.x * blockDim.x + threadIdx.x;
  const long stride = (long)gridDim.x * blockDim.x;
  for (; i < n; i += stride) d[i] = __float2bfloat16(s[i]);
}

__global__ void gather_rows(const bf16s* __restrict__ src,
                            const int* __restrict__ idx,
                            bf16s* __restrict__ dst, int cols, long n) {
  long i = (long)blockIdx.x * blockDim.x + threadIdx.x;
  const long stride = (long)gridDim.x * blockDim.x;
  for (; i < n; i += stride) {
    const int r = (int)(i / cols);
    const int c = (int)(i % cols);
    dst[i] = src[(size_t)idx[r] * cols + c];
  }
}

// q = rope(qkv[:, kv, g, :]) -> Qr bf16 (T, 32, 128)
__global__ void rope_q(const float* __restrict__ qkv,
                       const float* __restrict__ cs,
                       const float* __restrict__ sn,
                       bf16s* __restrict__ Qr) {
  long i = (long)blockIdx.x * blockDim.x + threadIdx.x;
  const long stride = (long)gridDim.x * blockDim.x;
  const long n = (long)T_SEQ * NHEADS * HD;
  for (; i < n; i += stride) {
    const int t = (int)(i >> 12);
    const int rem = (int)(i & 4095);
    const int h = rem >> 7, d = rem & 127;
    const int kv = h >> 2, g = h & 3;
    const size_t base = (size_t)t * QKV_OUT + kv * 768 + g * 128;
    const float x   = qkv[base + d];
    const float rot = (d < 64) ? -qkv[base + d + 64] : qkv[base + d - 64];
    Qr[i] = __float2bfloat16(x * cs[t * HD + d] + rot * sn[t * HD + d]);
  }
}

// k = rope(qkv slot 4) -> Kr bf16 (T, 8, 128);  v -> Vt bf16 (8, 128, T)
__global__ void rope_kv(const float* __restrict__ qkv,
                        const float* __restrict__ cs,
                        const float* __restrict__ sn,
                        bf16s* __restrict__ Kr, bf16s* __restrict__ Vt) {
  long i = (long)blockIdx.x * blockDim.x + threadIdx.x;
  const long stride = (long)gridDim.x * blockDim.x;
  const long n = (long)T_SEQ * NKV * HD;
  for (; i < n; i += stride) {
    const int t = (int)(i >> 10);
    const int rem = (int)(i & 1023);
    const int kv = rem >> 7, d = rem & 127;
    const size_t base = (size_t)t * QKV_OUT + kv * 768;
    const float k   = qkv[base + 512 + d];
    const float rot = (d < 64) ? -qkv[base + 512 + d + 64]
                               : -0.0f + qkv[base + 512 + d - 64];
    Kr[i] = __float2bfloat16(k * cs[t * HD + d] + rot * sn[t * HD + d]);
    Vt[(size_t)kv * HD * T_SEQ + (size_t)d * T_SEQ + t] =
        __float2bfloat16(qkv[base + 640 + d]);
  }
}

__global__ void cache_copy(const float* __restrict__ kc,
                           const float* __restrict__ vc,
                           float* __restrict__ ok, float* __restrict__ ov,
                           long n) {
  long i = (long)blockIdx.x * blockDim.x + threadIdx.x;
  const long stride = (long)gridDim.x * blockDim.x;
  for (; i < n; i += stride) { ok[i] = kc[i]; ov[i] = vc[i]; }
}

// fill_k = beacon_rope(pre-rope k[mem_idx]); fill_v = v[mem_idx]
__global__ void cache_fill(const float* __restrict__ qkv,
                           const float* __restrict__ bcs,
                           const float* __restrict__ bsn,
                           const int* __restrict__ mem_idx,
                           const int* __restrict__ boff,
                           float* __restrict__ ok, float* __restrict__ ov) {
  long i = (long)blockIdx.x * blockDim.x + threadIdx.x;
  const long stride = (long)gridDim.x * blockDim.x;
  const long n = (long)N_MEM * NKV * HD;
  for (; i < n; i += stride) {
    const int m = (int)(i >> 10);
    const int rem = (int)(i & 1023);
    const int kv = rem >> 7, d = rem & 127;
    const int t = mem_idx[m];
    const int blk = boff[m >> 6];
    const int off = m & 63;
    const size_t qb = (size_t)t * QKV_OUT + kv * 768;
    const float k   = qkv[qb + 512 + d];
    const float rot = (d < 64) ? -qkv[qb + 512 + d + 64] : qkv[qb + 512 + d - 64];
    const size_t oidx = (size_t)blk * (64 * NKV * HD) + (size_t)off * (NKV * HD)
                      + kv * HD + d;
    ok[oidx] = k * bcs[m * HD + d] + rot * bsn[m * HD + d];
    ov[oidx] = qkv[qb + 640 + d];
  }
}

// ---------------- Launcher ---------------------------------------------------

static int nb(long n, int thr) {
  long b = (n + thr - 1) / thr;
  return (int)(b > 16384 ? 16384 : b);
}

extern "C" void kernel_launch(void* const* d_in, const int* in_sizes, int n_in,
                              void* d_out, int out_size, void* d_ws, size_t ws_size,
                              hipStream_t stream) {
  const float* x        = (const float*)d_in[0];
  const float* cs       = (const float*)d_in[1];
  const float* sn       = (const float*)d_in[2];
  const float* bcs      = (const float*)d_in[3];
  const float* bsn      = (const float*)d_in[4];
  const int*   mem_idx  = (const int*)d_in[5];
  const int*   im_idx   = (const int*)d_in[6];
  const float* k_cache  = (const float*)d_in[7];
  const float* v_cache  = (const float*)d_in[8];
  const int*   boff     = (const int*)d_in[9];
  const float* wqkv_w   = (const float*)d_in[10];
  const float* wqkv_A   = (const float*)d_in[11];
  const float* wqkv_B   = (const float*)d_in[12];
  const float* wqkv_MA  = (const float*)d_in[13];
  const float* wqkv_MB  = (const float*)d_in[14];
  const float* wo_w     = (const float*)d_in[15];
  const float* wo_A     = (const float*)d_in[16];
  const float* wo_B     = (const float*)d_in[17];
  const float* wo_MA    = (const float*)d_in[18];
  const float* wo_MB    = (const float*)d_in[19];

  float* out  = (float*)d_out;                       // (2048, 4096) f32
  float* outK = out + (size_t)T_SEQ * H_DIM;         // (4,64,8,128) f32
  float* outV = outK + 4 * 64 * NKV * HD;

  // workspace carve-out (~230 MB)
  char* ws = (char*)d_ws;
  size_t off = 0;
  auto alloc = [&](size_t bytes) -> void* {
    off = (off + 255) & ~(size_t)255;
    void* p = ws + off;
    off += bytes;
    return p;
  };
  bf16s* Xb     = (bf16s*)alloc((size_t)T_SEQ * H_DIM * 2);
  bf16s* Wqkvb  = (bf16s*)alloc((size_t)QKV_OUT * H_DIM * 2);
  bf16s* Wob    = (bf16s*)alloc((size_t)H_DIM * H_DIM * 2);
  bf16s* qAb    = (bf16s*)alloc((size_t)LORA_R * H_DIM * 2);
  bf16s* qBb    = (bf16s*)alloc((size_t)QKV_OUT * LORA_R * 2);
  bf16s* qMAb   = (bf16s*)alloc((size_t)LORA_R * H_DIM * 2);
  bf16s* qMBb   = (bf16s*)alloc((size_t)QKV_OUT * LORA_R * 2);
  bf16s* oAb    = (bf16s*)alloc((size_t)LORA_R * H_DIM * 2);
  bf16s* oBb    = (bf16s*)alloc((size_t)H_DIM * LORA_R * 2);
  bf16s* oMAb   = (bf16s*)alloc((size_t)LORA_R * H_DIM * 2);
  bf16s* oMBb   = (bf16s*)alloc((size_t)H_DIM * LORA_R * 2);
  float* qkv    = (float*)alloc((size_t)T_SEQ * QKV_OUT * 4);
  bf16s* Xim    = (bf16s*)alloc((size_t)N_IM * H_DIM * 2);
  bf16s* Xmem   = (bf16s*)alloc((size_t)N_MEM * H_DIM * 2);
  bf16s* Uim    = (bf16s*)alloc((size_t)N_IM * LORA_R * 2);
  bf16s* Umem   = (bf16s*)alloc((size_t)N_MEM * LORA_R * 2);
  bf16s* Qr     = (bf16s*)alloc((size_t)T_SEQ * NHEADS * HD * 2);
  bf16s* Kr     = (bf16s*)alloc((size_t)T_SEQ * NKV * HD * 2);
  bf16s* Vt     = (bf16s*)alloc((size_t)NKV * HD * T_SEQ * 2);
  bf16s* Attnb  = (bf16s*)alloc((size_t)T_SEQ * H_DIM * 2);
  bf16s* Aim    = (bf16s*)alloc((size_t)N_IM * H_DIM * 2);
  bf16s* Amem   = (bf16s*)alloc((size_t)N_MEM * H_DIM * 2);
  (void)ws_size; (void)in_sizes; (void)n_in; (void)out_size;

  const int THR = 256;
  // ---- casts to bf16 ----
  cast_f32_to_bf16<<<nb((long)T_SEQ * H_DIM, THR), THR, 0, stream>>>(x, Xb, (long)T_SEQ * H_DIM);
  cast_f32_to_bf16<<<nb((long)QKV_OUT * H_DIM, THR), THR, 0, stream>>>(wqkv_w, Wqkvb, (long)QKV_OUT * H_DIM);
  cast_f32_to_bf16<<<nb((long)H_DIM * H_DIM, THR), THR, 0, stream>>>(wo_w, Wob, (long)H_DIM * H_DIM);
  cast_f32_to_bf16<<<nb((long)LORA_R * H_DIM, THR), THR, 0, stream>>>(wqkv_A, qAb, (long)LORA_R * H_DIM);
  cast_f32_to_bf16<<<nb((long)QKV_OUT * LORA_R, THR), THR, 0, stream>>>(wqkv_B, qBb, (long)QKV_OUT * LORA_R);
  cast_f32_to_bf16<<<nb((long)LORA_R * H_DIM, THR), THR, 0, stream>>>(wqkv_MA, qMAb, (long)LORA_R * H_DIM);
  cast_f32_to_bf16<<<nb((long)QKV_OUT * LORA_R, THR), THR, 0, stream>>>(wqkv_MB, qMBb, (long)QKV_OUT * LORA_R);
  cast_f32_to_bf16<<<nb((long)LORA_R * H_DIM, THR), THR, 0, stream>>>(wo_A, oAb, (long)LORA_R * H_DIM);
  cast_f32_to_bf16<<<nb((long)H_DIM * LORA_R, THR), THR, 0, stream>>>(wo_B, oBb, (long)H_DIM * LORA_R);
  cast_f32_to_bf16<<<nb((long)LORA_R * H_DIM, THR), THR, 0, stream>>>(wo_MA, oMAb, (long)LORA_R * H_DIM);
  cast_f32_to_bf16<<<nb((long)H_DIM * LORA_R, THR), THR, 0, stream>>>(wo_MB, oMBb, (long)H_DIM * LORA_R);

  // ---- qkv = x @ Wqkv^T ----
  gemm_bf16_wmma<0><<<dim3(T_SEQ / 128, QKV_OUT / 256), 256, 0, stream>>>(
      Xb, Wqkvb, qkv, nullptr, H_DIM, H_DIM, H_DIM, QKV_OUT, nullptr, 0.f);

  // ---- qkv LoRA (im + mem) ----
  gather_rows<<<nb((long)N_IM * H_DIM, THR), THR, 0, stream>>>(Xb, im_idx, Xim, H_DIM, (long)N_IM * H_DIM);
  gather_rows<<<nb((long)N_MEM * H_DIM, THR), THR, 0, stream>>>(Xb, mem_idx, Xmem, H_DIM, (long)N_MEM * H_DIM);
  gemm_bf16_wmma<1><<<dim3(N_IM / 128, LORA_R / 256), 256, 0, stream>>>(
      Xim, qAb, nullptr, Uim, H_DIM, H_DIM, H_DIM, LORA_R, nullptr, 0.f);
  gemm_bf16_wmma<2><<<dim3(N_IM / 128, QKV_OUT / 256), 256, 0, stream>>>(
      Uim, qBb, qkv, nullptr, LORA_R, LORA_R, LORA_R, QKV_OUT, im_idx, 1.0f);
  gemm_bf16_wmma<1><<<dim3(N_MEM / 128, LORA_R / 256), 256, 0, stream>>>(
      Xmem, qMAb, nullptr, Umem, H_DIM, H_DIM, H_DIM, LORA_R, nullptr, 0.f);
  gemm_bf16_wmma<2><<<dim3(N_MEM / 128, QKV_OUT / 256), 256, 0, stream>>>(
      Umem, qMBb, qkv, nullptr, LORA_R, LORA_R, LORA_R, QKV_OUT, mem_idx, 1.0f);

  // ---- KV cache outputs (pre-rope k, beacon rope) ----
  cache_copy<<<nb(4L * 64 * NKV * HD, THR), THR, 0, stream>>>(
      k_cache, v_cache, outK, outV, 4L * 64 * NKV * HD);
  cache_fill<<<nb((long)N_MEM * NKV * HD, THR), THR, 0, stream>>>(
      qkv, bcs, bsn, mem_idx, boff, outK, outV);

  // ---- RoPE + layouts for attention ----
  rope_q<<<nb((long)T_SEQ * NHEADS * HD, THR), THR, 0, stream>>>(qkv, cs, sn, Qr);
  rope_kv<<<nb((long)T_SEQ * NKV * HD, THR), THR, 0, stream>>>(qkv, cs, sn, Kr, Vt);

  // ---- causal GQA flash attention ----
  attn_fwd<<<dim3(T_SEQ / 16, NHEADS), 32, 0, stream>>>(Qr, Kr, Vt, Attnb);

  // ---- out = attn @ Wo^T ----
  gemm_bf16_wmma<0><<<dim3(T_SEQ / 128, H_DIM / 256), 256, 0, stream>>>(
      Attnb, Wob, out, nullptr, H_DIM, H_DIM, H_DIM, H_DIM, nullptr, 0.f);

  // ---- wo LoRA (im + mem) ----
  gather_rows<<<nb((long)N_IM * H_DIM, THR), THR, 0, stream>>>(Attnb, im_idx, Aim, H_DIM, (long)N_IM * H_DIM);
  gather_rows<<<nb((long)N_MEM * H_DIM, THR), THR, 0, stream>>>(Attnb, mem_idx, Amem, H_DIM, (long)N_MEM * H_DIM);
  gemm_bf16_wmma<1><<<dim3(N_IM / 128, LORA_R / 256), 256, 0, stream>>>(
      Aim, oAb, nullptr, Uim, H_DIM, H_DIM, H_DIM, LORA_R, nullptr, 0.f);
  gemm_bf16_wmma<2><<<dim3(N_IM / 128, H_DIM / 256), 256, 0, stream>>>(
      Uim, oBb, out, nullptr, LORA_R, LORA_R, LORA_R, H_DIM, im_idx, 1.0f);
  gemm_bf16_wmma<1><<<dim3(N_MEM / 128, LORA_R / 256), 256, 0, stream>>>(
      Amem, oMAb, nullptr, Umem, H_DIM, H_DIM, H_DIM, LORA_R, nullptr, 0.f);
  gemm_bf16_wmma<2><<<dim3(N_MEM / 128, H_DIM / 256), 256, 0, stream>>>(
      Umem, oMBb, out, nullptr, LORA_R, LORA_R, LORA_R, H_DIM, mem_idx, 1.0f);
}
